// Attention_51634096833229
// MI455X (gfx1250) — matrix operand order repaired and checked by reading, not compiled, not run on hardware
//
#include <hip/hip_runtime.h>
#include <hip/hip_bf16.h>

typedef _Float16 v16h __attribute__((ext_vector_type(16)));
typedef _Float16 v8h  __attribute__((ext_vector_type(8)));
typedef float    v8f  __attribute__((ext_vector_type(8)));

#define NB   32
#define TT   785
#define CC   192
#define HD   64
#define MROWS (NB*TT)          // 25120, divisible by 16
#define SCALE 0.07216878365f   // 192^-0.5 (reference uses Co, not d)

// ---------------------------------------------------------------------------
// Kernel 0: one-time f32 -> f16 conversion of w_qkv (3*192*192) and w_proj.
// ---------------------------------------------------------------------------
__global__ void cvt_weights(const float* __restrict__ wq,
                            const float* __restrict__ wp,
                            _Float16* __restrict__ whq,
                            _Float16* __restrict__ whp) {
  int i = blockIdx.x * blockDim.x + threadIdx.x;
  if (i < 3 * CC * CC) whq[i] = (_Float16)wq[i];
  if (i < CC * CC)     whp[i] = (_Float16)wp[i];
}

// ---------------------------------------------------------------------------
// Kernel 1: depthwise 3x3 conv + BN for all 3 branches; cls token copied.
// y layout: [branch][b][t][c], f16.
// ---------------------------------------------------------------------------
__global__ void convbn_kernel(const float* __restrict__ x,
                              const float* __restrict__ cw,
                              const float* __restrict__ g,
                              const float* __restrict__ bb,
                              const float* __restrict__ mu,
                              const float* __restrict__ var,
                              _Float16* __restrict__ y) {
  long idx = (long)blockIdx.x * blockDim.x + threadIdx.x;
  const long total = 3L * NB * TT * CC;
  if (idx >= total) return;
  int c = (int)(idx % CC);
  long r = idx / CC;
  int t = (int)(r % TT); r /= TT;
  int b = (int)(r % NB);
  int i = (int)(r / NB);
  const float* xb = x + (long)b * TT * CC;
  float out;
  if (t == 0) {
    out = xb[c];  // cls token passes through untouched
  } else {
    int hh = (t - 1) / 28, ww = (t - 1) % 28;
    float s    = g[i*CC + c] * rsqrtf(var[i*CC + c] + 1e-5f);
    float bias = bb[i*CC + c] - mu[i*CC + c] * s;
    const float* w9 = cw + ((long)i * CC + c) * 9;
    float acc = 0.0f;
#pragma unroll
    for (int kh = 0; kh < 3; ++kh) {
#pragma unroll
      for (int kw = 0; kw < 3; ++kw) {
        int ih = hh + kh - 1, iw = ww + kw - 1;
        if (ih >= 0 && ih < 28 && iw >= 0 && iw < 28)
          acc += xb[(1 + ih*28 + iw) * CC + c] * w9[kh*3 + kw];
      }
    }
    out = acc * s + bias;
  }
  y[idx] = (_Float16)out;
}

// ---------------------------------------------------------------------------
// Kernel 2: WMMA GEMM  out[m,n] = sum_k A[m,k] * Wh[n,k] (+bias)
// A: f16 row-major MxK (K=192). Wh: f16 NxK (pre-converted).
// One wave computes a 16x64 output strip (4 n-tiles share one A tile).
// Loads for all 4 B tiles are issued before the 4 WMMAs so the backend can
// clause them and stage the loadcnt waits instead of wait-0 per WMMA.
// grid = (M/16, N/64, nbranch), block = 32 (one wave; EXEC all-1s at WMMA).
// ---------------------------------------------------------------------------
__global__ void __launch_bounds__(32)
gemm_wmma(const _Float16* __restrict__ Aall,
          const _Float16* __restrict__ Wall,
          const float* __restrict__ bias,
          _Float16* __restrict__ outH,
          float* __restrict__ outF,
          int M, int N, int K) {
  const int lane = threadIdx.x & 31;
  const int hf = lane >> 4, l16 = lane & 15;
  const int mt = blockIdx.x, nq = blockIdx.y, z = blockIdx.z;
  const _Float16* A = Aall + (size_t)z * M * K;
  const _Float16* W = Wall + (size_t)z * N * K;
  const size_t arow = (size_t)(mt*16 + l16) * K;   // A-layout: M = lane%16
  const int n0 = nq * 64;

  v8f acc[4];
#pragma unroll
  for (int nt = 0; nt < 4; ++nt)
#pragma unroll
    for (int e = 0; e < 8; ++e) acc[nt][e] = 0.0f;

  for (int k0 = 0; k0 < K; k0 += 32) {
    // prefetch next A k-chunk (speculative; OOB prefetch is dropped)
    __builtin_prefetch(A + arow + k0 + 32, 0, 1);
    // A tile, 16-bit A-layout: lane half h holds K = k0+16h..+15 (contiguous, matches B)
    v8h alo = *(const v8h*)(A + arow + k0 + hf*16);
    v8h ahi = *(const v8h*)(A + arow + k0 + hf*16 + 8);
    // 4 B tiles: lanes 0-15 K=k0..k0+15, lanes 16-31 K=k0+16..+31; N = lane%16
    v16h bm[4];
#pragma unroll
    for (int nt = 0; nt < 4; ++nt)
      bm[nt] = *(const v16h*)(W + (size_t)(n0 + nt*16 + l16) * K + k0 + hf*16);
    v16h a;
#pragma unroll
    for (int e = 0; e < 8; ++e) { a[e] = alo[e]; a[e+8] = ahi[e]; }
#pragma unroll
    for (int nt = 0; nt < 4; ++nt)
      acc[nt] = __builtin_amdgcn_wmma_f32_16x16x32_f16(false, a, false, bm[nt],
                                                (short)0, acc[nt], false, false);
  }
  // C/D layout: VGPR r, lane group hf -> row = r + 8*hf ; col = lane%16
  if (outF) {
#pragma unroll
    for (int nt = 0; nt < 4; ++nt) {
      const int n = n0 + nt*16 + l16;
      const float bv = bias ? bias[n] : 0.0f;
#pragma unroll
      for (int r = 0; r < 8; ++r)
        outF[(size_t)(mt*16 + hf*8 + r) * N + n] = acc[nt][r] + bv;
    }
  } else {
    _Float16* o = outH + (size_t)z * M * N;
#pragma unroll
    for (int nt = 0; nt < 4; ++nt) {
      const int n = n0 + nt*16 + l16;
#pragma unroll
      for (int r = 0; r < 8; ++r)
        o[(size_t)(mt*16 + hf*8 + r) * N + n] = (_Float16)acc[nt][r];
    }
  }
}

// ---------------------------------------------------------------------------
// Kernel 3: flash attention. One wave per (16-query tile, head, batch).
// qkv: [3][b][t][192] f16.  aout: [b][t][192] f16.
// ---------------------------------------------------------------------------
__global__ void __launch_bounds__(32)
attn_kernel(const _Float16* __restrict__ qkv, _Float16* __restrict__ aout) {
  const int lane = threadIdx.x & 31;
  const int hf = lane >> 4, l16 = lane & 15;
  const int qt = blockIdx.x;   // 0..49
  const int hd = blockIdx.y;   // 0..2
  const int b  = blockIdx.z;   // 0..31
  const size_t per = (size_t)NB * TT * CC;
  const _Float16* Q  = qkv;
  const _Float16* Km = qkv + per;
  const _Float16* V  = qkv + 2*per;

  __shared__ _Float16 P[16 * 34];  // 16x32 P chunk, stride 34 (bank-friendly)

  // ---- load q tile (16 x 64) into two A-layout register tiles, K=0..63
  int tq = qt*16 + l16; if (tq > TT-1) tq = TT-1;
  const _Float16* qrow = Q + ((size_t)b*TT + tq)*CC + hd*HD;
  v16h a0, a1;
  {
    v8h lo = *(const v8h*)(qrow + hf*16);
    v8h hi = *(const v8h*)(qrow + hf*16 + 8);
#pragma unroll
    for (int e = 0; e < 8; ++e) { a0[e] = lo[e]; a0[e+8] = hi[e]; }
    lo = *(const v8h*)(qrow + 32 + hf*16);
    hi = *(const v8h*)(qrow + 40 + hf*16);
#pragma unroll
    for (int e = 0; e < 8; ++e) { a1[e] = lo[e]; a1[e+8] = hi[e]; }
  }

  float mrun[8], lrun[8];
  v8f accO[4];
#pragma unroll
  for (int r = 0; r < 8; ++r) { mrun[r] = -3.0e38f; lrun[r] = 0.0f; }
#pragma unroll
  for (int nt = 0; nt < 4; ++nt)
#pragma unroll
    for (int e = 0; e < 8; ++e) accO[nt][e] = 0.0f;

  for (int ch = 0; ch < 25; ++ch) {          // 25 chunks x 32 keys = 800
    v8f s[2];
#pragma unroll
    for (int sub = 0; sub < 2; ++sub) {
      int key = ch*32 + sub*16 + l16;        // this lane's key column
      int tk = key > TT-1 ? TT-1 : key;
      const _Float16* krow = Km + ((size_t)b*TT + tk)*CC + hd*HD;
      v16h b0 = *(const v16h*)(krow + hf*16);        // dims 0..31
      v16h b1 = *(const v16h*)(krow + 32 + hf*16);   // dims 32..63
      v8f sv;
#pragma unroll
      for (int e = 0; e < 8; ++e) sv[e] = 0.0f;
      sv = __builtin_amdgcn_wmma_f32_16x16x32_f16(false, a0, false, b0,
                                                  (short)0, sv, false, false);
      sv = __builtin_amdgcn_wmma_f32_16x16x32_f16(false, a1, false, b1,
                                                  (short)0, sv, false, false);
      bool valid = key < TT;
#pragma unroll
      for (int r = 0; r < 8; ++r) s[sub][r] = valid ? sv[r]*SCALE : -3.0e38f;
    }
    // ---- online softmax update (row reductions across 16 lanes per half)
    float f[8];
#pragma unroll
    for (int r = 0; r < 8; ++r) {
      float mx = fmaxf(s[0][r], s[1][r]);
#pragma unroll
      for (int off = 1; off < 16; off <<= 1)
        mx = fmaxf(mx, __shfl_xor(mx, off, 16));
      float mn = fmaxf(mrun[r], mx);
      float p0 = __expf(s[0][r] - mn);
      float p1 = __expf(s[1][r] - mn);
      float rs = p0 + p1;
#pragma unroll
      for (int off = 1; off < 16; off <<= 1)
        rs += __shfl_xor(rs, off, 16);
      f[r] = __expf(mrun[r] - mn);
      lrun[r] = lrun[r]*f[r] + rs;
      mrun[r] = mn;
      P[(r + hf*8)*34 + l16]      = (_Float16)p0;   // C-layout -> LDS
      P[(r + hf*8)*34 + 16 + l16] = (_Float16)p1;
    }
#pragma unroll
    for (int nt = 0; nt < 4; ++nt)
#pragma unroll
      for (int r = 0; r < 8; ++r) accO[nt][r] *= f[r];
    __syncthreads();
    // ---- reload P in A-layout (row = lane%16, K = keys in chunk)
    v16h pa;
    {
      const _Float16* pr = P + l16*34;
#pragma unroll
      for (int e = 0; e < 8; ++e) {
        pa[e]   = pr[hf*16 + e];
        pa[e+8] = pr[hf*16 + 8 + e];
      }
    }
    // ---- O += P @ V: load all 4 V B-tiles first, then the 4 WMMAs
    v16h bv[4];
#pragma unroll
    for (int nt = 0; nt < 4; ++nt) {
#pragma unroll
      for (int e = 0; e < 16; ++e) {
        int key = ch*32 + hf*16 + e;
        int tv = key > TT-1 ? TT-1 : key;
        bv[nt][e] = V[((size_t)b*TT + tv)*CC + hd*HD + nt*16 + l16];
      }
    }
#pragma unroll
    for (int nt = 0; nt < 4; ++nt)
      accO[nt] = __builtin_amdgcn_wmma_f32_16x16x32_f16(false, pa, false, bv[nt],
                                                (short)0, accO[nt], false, false);
    __syncthreads();
  }
  // ---- epilogue: divide by l, write valid rows
#pragma unroll
  for (int r = 0; r < 8; ++r) {
    int t = qt*16 + hf*8 + r;
    if (t < TT) {
      float inv = 1.0f / lrun[r];
#pragma unroll
      for (int nt = 0; nt < 4; ++nt)
        aout[((size_t)b*TT + t)*CC + hd*HD + nt*16 + l16] =
            (_Float16)(accO[nt][r] * inv);
    }
  }
}

// ---------------------------------------------------------------------------
extern "C" void kernel_launch(void* const* d_in, const int* in_sizes, int n_in,
                              void* d_out, int out_size, void* d_ws, size_t ws_size,
                              hipStream_t stream) {
  const float* x        = (const float*)d_in[0];
  const float* conv_w   = (const float*)d_in[1];
  const float* bn_scale = (const float*)d_in[2];
  const float* bn_bias  = (const float*)d_in[3];
  const float* bn_mean  = (const float*)d_in[4];
  const float* bn_var   = (const float*)d_in[5];
  const float* w_qkv    = (const float*)d_in[6];
  const float* w_proj   = (const float*)d_in[7];
  const float* b_proj   = (const float*)d_in[8];
  float* out = (float*)d_out;

  const size_t per = (size_t)NB * TT * CC;   // 4,823,040 elems
  _Float16* y    = (_Float16*)d_ws;          // 3 branches f16
  _Float16* qkvb = y + 3*per;                // q,k,v f16
  _Float16* aout = qkvb + 3*per;             // attention output f16
  _Float16* whq  = aout + per;               // f16 w_qkv  (3*192*192)
  _Float16* whp  = whq + 3*CC*CC;            // f16 w_proj (192*192)

  // 0) weight conversion (one-shot, 147456 elements)
  cvt_weights<<<dim3((3*CC*CC + 255) / 256), 256, 0, stream>>>(
      w_qkv, w_proj, whq, whp);

  // 1) depthwise conv + BN (3 branches)
  long total = 3L * (long)per;
  convbn_kernel<<<dim3((unsigned)((total + 255) / 256)), 256, 0, stream>>>(
      x, conv_w, bn_scale, bn_bias, bn_mean, bn_var, y);

  // 2) QKV projections: 3 x (25120x192)·(192x192), 16x64 strip per wave
  gemm_wmma<<<dim3(MROWS/16, CC/64, 3), 32, 0, stream>>>(
      y, whq, nullptr, qkvb, nullptr, MROWS, CC, CC);

  // 3) flash attention: 50 query tiles x 3 heads x 32 batches
  attn_kernel<<<dim3(50, 3, NB), 32, 0, stream>>>(qkvb, aout);

  // 4) output projection + bias -> f32 d_out
  gemm_wmma<<<dim3(MROWS/16, CC/64, 1), 32, 0, stream>>>(
      aout, whp, b_proj, nullptr, out, MROWS, CC, CC);
}
